// TimeFinNLPCL_55044300865978
// MI455X (gfx1250) — matrix operand order repaired and checked by reading, not compiled, not run on hardware
//
#include <hip/hip_runtime.h>

// ---------------------------------------------------------------------------
// Types for CDNA5 WMMA (wave32): V_WMMA_F32_16X16X32_BF16
// ---------------------------------------------------------------------------
typedef __bf16 bf16;
typedef __attribute__((ext_vector_type(8)))  __bf16 v8bf;
typedef __attribute__((ext_vector_type(16))) __bf16 v16bf;
typedef __attribute__((ext_vector_type(8)))  float  v8f;

#define DEV static __device__ __forceinline__

// Problem constants (from reference)
#define BN   256   // batch
#define DD   5     // days
#define SS   30    // seq len
#define EE   768   // embed
#define H1   512
#define H2   512
#define G1   2048  // 4*H1
#define FF   256
#define OUTD 2

DEV float sigmoidf_(float x) { return 1.0f / (1.0f + __expf(-x)); }

DEV v16bf cat8(v8bf lo, v8bf hi) {
  return __builtin_shufflevector(lo, hi, 0,1,2,3,4,5,6,7,8,9,10,11,12,13,14,15);
}

// A fragment: 16x32 bf16, row-major source. Lane l<16 -> row l, K {k..k+7, k+16..k+23}
// Lane l>=16 -> row l-16, K {k+8..k+15, k+24..k+31}. Two 16B loads per lane.
DEV v16bf load_a_frag(const bf16* __restrict__ A, int lda, int m0, int k, int lane) {
  const bf16* p = A + (size_t)(m0 + (lane & 15)) * lda + k + ((lane >> 4) << 3);
  v8bf lo = *(const v8bf*)(p);
  v8bf hi = *(const v8bf*)(p + 16);
  __builtin_prefetch((const void*)(p + 32), 0, 1);   // global_prefetch_b8
  return cat8(lo, hi);
}

// B fragment: 32x16 (KxN) where memory holds W as (N,K) row-major (torch layout).
// Lane l<16 -> col l, K {k..k+7, k+16..k+23}; lane l>=16 -> col l-16, K {k+8..k+15, k+24..k+31}.
DEV v16bf load_b_frag(const bf16* __restrict__ B, int ldb, int n0, int k, int lane) {
  const bf16* p = B + (size_t)(n0 + (lane & 15)) * ldb + k + ((lane >> 4) << 3);
  v8bf lo = *(const v8bf*)(p);
  v8bf hi = *(const v8bf*)(p + 16);
  __builtin_prefetch((const void*)(p + 32), 0, 1);
  return cat8(lo, hi);
}

// ---------------------------------------------------------------------------
// Generic bf16 WMMA GEMM: C[M,N] = act( A[M,K] @ W[N,K]^T + bias + addend )
// A may be split along K (A1: K1 cols, A2: K2 cols) to fuse [h | x] @ [Wall|Uall]^T.
// Score mode (score != nullptr): instead of storing C, computes
//   atomicAdd(score[m], sum_n V[n]*tanh(acc + bias[n] + dproj[m/drowdiv][n]))
// Block = 128 threads (4 waves). Wave tile = 16(M) x 64(N): one A fragment is
// reused by 4 WMMAs. All 5 fragments of a k-step are loaded into distinct live
// registers BEFORE any WMMA issues, so the B loads for tiles 1..3 (and the
// next k-step's loads) overlap with WMMA execution (partial s_wait_loadcnt).
// Waves stack along M: block tile 64(M) x 64(N). Grid = (N/64, M/64).
// All M multiples of 64, N multiples of 64, K multiples of 32 in this model.
// ---------------------------------------------------------------------------
__global__ __launch_bounds__(128)
void k_gemm(const bf16* __restrict__ A1, int lda1, int K1,
            const bf16* __restrict__ A2, int lda2, int K2,
            const bf16* __restrict__ Bw, int ldb,
            const float* __restrict__ bias,
            const float* __restrict__ addend, int add_stride,
            float* __restrict__ C, int ldc, int act,
            const float* __restrict__ vvec,
            const float* __restrict__ dproj, int drowdiv,
            float* __restrict__ score)
{
  const int lane = threadIdx.x & 31;
  const int wave = threadIdx.x >> 5;
  const int n0 = blockIdx.x * 64;
  const int m0 = blockIdx.y * 64 + wave * 16;

  v8f acc0 = (v8f){0.f,0.f,0.f,0.f,0.f,0.f,0.f,0.f};
  v8f acc1 = acc0, acc2 = acc0, acc3 = acc0;

  for (int k = 0; k < K1; k += 32) {
    // load all fragments first: 10 b128 loads in flight before any math
    v16bf a  = load_a_frag(A1, lda1, m0, k, lane);
    v16bf b0 = load_b_frag(Bw, ldb, n0 +  0, k, lane);
    v16bf b1 = load_b_frag(Bw, ldb, n0 + 16, k, lane);
    v16bf b2 = load_b_frag(Bw, ldb, n0 + 32, k, lane);
    v16bf b3 = load_b_frag(Bw, ldb, n0 + 48, k, lane);
    acc0 = __builtin_amdgcn_wmma_f32_16x16x32_bf16(false, a, false, b0, (short)0, acc0, false, false);
    acc1 = __builtin_amdgcn_wmma_f32_16x16x32_bf16(false, a, false, b1, (short)0, acc1, false, false);
    acc2 = __builtin_amdgcn_wmma_f32_16x16x32_bf16(false, a, false, b2, (short)0, acc2, false, false);
    acc3 = __builtin_amdgcn_wmma_f32_16x16x32_bf16(false, a, false, b3, (short)0, acc3, false, false);
  }
  for (int k = 0; k < K2; k += 32) {
    v16bf a  = load_a_frag(A2, lda2, m0, k, lane);
    v16bf b0 = load_b_frag(Bw, ldb, n0 +  0, K1 + k, lane);
    v16bf b1 = load_b_frag(Bw, ldb, n0 + 16, K1 + k, lane);
    v16bf b2 = load_b_frag(Bw, ldb, n0 + 32, K1 + k, lane);
    v16bf b3 = load_b_frag(Bw, ldb, n0 + 48, K1 + k, lane);
    acc0 = __builtin_amdgcn_wmma_f32_16x16x32_bf16(false, a, false, b0, (short)0, acc0, false, false);
    acc1 = __builtin_amdgcn_wmma_f32_16x16x32_bf16(false, a, false, b1, (short)0, acc1, false, false);
    acc2 = __builtin_amdgcn_wmma_f32_16x16x32_bf16(false, a, false, b2, (short)0, acc2, false, false);
    acc3 = __builtin_amdgcn_wmma_f32_16x16x32_bf16(false, a, false, b3, (short)0, acc3, false, false);
  }

  v8f accs[4] = {acc0, acc1, acc2, acc3};

  if (score == nullptr) {
    #pragma unroll
    for (int t = 0; t < 4; ++t) {
      const int nn = n0 + 16 * t + (lane & 15);
      const float bv = bias ? bias[nn] : 0.0f;
      #pragma unroll
      for (int r = 0; r < 8; ++r) {
        int mm = m0 + r + ((lane >> 4) << 3);
        float v = accs[t][r] + bv;
        if (addend) v += addend[(size_t)mm * add_stride + nn];
        if (act == 1) v = tanhf(v);
        else if (act == 2) v = fmaxf(v, 0.0f);
        C[(size_t)mm * ldc + nn] = v;
      }
    }
  } else {
    #pragma unroll
    for (int r = 0; r < 8; ++r) {
      int mm = m0 + r + ((lane >> 4) << 3);
      const float* dp = dproj + (size_t)(mm / drowdiv) * 512;
      float v = 0.0f;
      #pragma unroll
      for (int t = 0; t < 4; ++t) {
        int nn = n0 + 16 * t + (lane & 15);
        float x = accs[t][r] + (bias ? bias[nn] : 0.0f) + dp[nn];
        v += vvec[nn] * tanhf(x);
      }
      // reduce across the 16 lanes holding row mm (xor 1,2,4,8 stays in half)
      v += __shfl_xor(v, 1);
      v += __shfl_xor(v, 2);
      v += __shfl_xor(v, 4);
      v += __shfl_xor(v, 8);
      if ((lane & 15) == 0) atomicAdd(&score[mm], v);
    }
  }
}

// ---------------------------------------------------------------------------
// Small utility kernels
// ---------------------------------------------------------------------------
__global__ void k_cvt(bf16* __restrict__ d, const float* __restrict__ s, int n) {
  int i = blockIdx.x * blockDim.x + threadIdx.x;
  if (i < n) d[i] = (bf16)s[i];
}

// x: sentence_feats (B,D,S,E) fp32  ->  xbf row r=(d*B+b)*S+s, col e  (38400 x 768)
__global__ void k_cvt_x(bf16* __restrict__ d, const float* __restrict__ s) {
  int i = blockIdx.x * blockDim.x + threadIdx.x;
  const int total = BN * DD * SS * EE;
  if (i >= total) return;
  int e = i % EE;
  int r = i / EE;
  int st = r % SS;
  int n  = r / SS;        // n = dd*B + b
  int dd = n >> 8;        // /256
  int b  = n & 255;
  d[i] = (bf16)s[(((size_t)(b * DD + dd) * SS) + st) * EE + e];
}

// Wcat (2048 x 1280) bf16 : [Wall (2048x512) | Uall (2048x768)]
__global__ void k_build_wcat(bf16* __restrict__ d, const float* __restrict__ wall,
                             const float* __restrict__ uall) {
  int i = blockIdx.x * blockDim.x + threadIdx.x;
  const int total = G1 * (H1 + EE);
  if (i >= total) return;
  int k = i % (H1 + EE);
  int n = i / (H1 + EE);
  float v = (k < H1) ? wall[(size_t)n * H1 + k] : uall[(size_t)n * EE + (k - H1)];
  d[i] = (bf16)v;
}

__global__ void k_add2(float* __restrict__ d, const float* __restrict__ a,
                       const float* __restrict__ b, int n) {
  int i = blockIdx.x * blockDim.x + threadIdx.x;
  if (i < n) d[i] = a[i] + b[i];
}

__global__ void k_zero_f32(float* p, int n) {
  int i = blockIdx.x * blockDim.x + threadIdx.x;
  if (i < n) p[i] = 0.0f;
}
__global__ void k_zero_bf16(bf16* p, int n) {
  int i = blockIdx.x * blockDim.x + threadIdx.x;
  if (i < n) p[i] = (bf16)0.0f;
}
__global__ void k_fill(float* p, int n, const float* __restrict__ v) {
  int i = blockIdx.x * blockDim.x + threadIdx.x;
  if (i < n) p[i] = v[0];
}

// ---------------------------------------------------------------------------
// TimeLSTM elementwise step. gates order (source): f, i, o, c_tmp (all sigmoid).
// c_s1 already tanh'd by the GEMM epilogue.  outs stores the o-gate (as source).
// ---------------------------------------------------------------------------
__global__ void k_tstep(const float* __restrict__ gates, const float* __restrict__ cs1,
                        float* __restrict__ c, bf16* __restrict__ c_bf,
                        bf16* __restrict__ h_bf,
                        float* __restrict__ outs, bf16* __restrict__ outs_bf,
                        const float* __restrict__ timef, int s)
{
  int i = blockIdx.x * blockDim.x + threadIdx.x;
  const int total = DD * BN * H1;          // 1280 * 512
  if (i >= total) return;
  int n = i >> 9;                          // row (d*256+b)
  int j = i & 511;
  int dd = n >> 8;
  int b  = n & 255;
  float ts   = timef[(b * DD + dd) * SS + s];
  float c1   = cs1[i];
  float cadj = c[i] - c1 + c1 * ts;
  const float* g = gates + (size_t)n * G1;
  float fg = sigmoidf_(g[j]);
  float ig = sigmoidf_(g[H1 + j]);
  float og = sigmoidf_(g[2 * H1 + j]);
  float ct = sigmoidf_(g[3 * H1 + j]);
  float cn = fg * cadj + ig * ct;
  float h  = og * tanhf(cn);
  c[i]    = cn;
  c_bf[i] = (bf16)cn;
  h_bf[i] = (bf16)h;
  size_t oi = ((size_t)n * SS + s) * H1 + j;
  outs[oi]    = og;
  outs_bf[oi] = (bf16)og;
}

// last_proj[d][j] = A1_W1_b[j] + sum_h A1_W1_w[j,h] * outs[d, B-1, idx_d, h]
__global__ void k_last_proj(float* __restrict__ lp, const float* __restrict__ outs,
                            const int* __restrict__ lt,
                            const float* __restrict__ w, const float* __restrict__ bias)
{
  int g = blockIdx.x * blockDim.x + threadIdx.x;
  if (g >= DD * H1) return;
  int dd = g >> 9;
  int j  = g & 511;
  int idx = lt[(BN - 1) * DD + dd] - 1;
  const float* last = outs + (((size_t)(dd * BN + (BN - 1)) * SS) + idx) * H1;
  float acc = bias[j];
  for (int h = 0; h < H1; ++h) acc += w[(size_t)j * H1 + h] * last[h];
  lp[g] = acc;
}

// attn1: masked softmax over S then weighted sum of outs -> lstm1_out (B,D,H1)
__global__ void k_attn1(const float* __restrict__ score, const float* __restrict__ outs,
                        const int* __restrict__ lt,
                        float* __restrict__ l1, bf16* __restrict__ l1_bf)
{
  __shared__ float aw[SS];
  int n  = blockIdx.x;            // d*256+b
  int dd = n >> 8;
  int b  = n & 255;
  int len = lt[b * DD + dd];
  if (threadIdx.x == 0) {
    const float* sc = score + (size_t)n * SS;
    float mx = -1e30f;
    for (int s = 0; s < len; ++s) mx = fmaxf(mx, sc[s]);
    float sum = 0.0f;
    for (int s = 0; s < len; ++s) { float e = __expf(sc[s] - mx); aw[s] = e; sum += e; }
    float inv = 1.0f / sum;
    for (int s = 0; s < len; ++s) aw[s] *= inv;
    for (int s = len; s < SS; ++s) aw[s] = 0.0f;
  }
  __syncthreads();
  for (int j = threadIdx.x; j < H1; j += blockDim.x) {
    float acc = 0.0f;
    for (int s = 0; s < SS; ++s) acc += aw[s] * outs[((size_t)n * SS + s) * H1 + j];
    size_t r = ((size_t)b * DD + dd) * H1 + j;   // (B,D,H1)
    l1[r]    = acc;
    l1_bf[r] = (bf16)acc;
  }
}

// std LSTM elementwise step. gates order: i, f, g, o.
__global__ void k_l2step(const float* __restrict__ g2, float* __restrict__ c2,
                         bf16* __restrict__ h2_bf,
                         float* __restrict__ hs, bf16* __restrict__ hs_bf, int t)
{
  int i = blockIdx.x * blockDim.x + threadIdx.x;
  const int total = BN * H2;               // 256*512
  if (i >= total) return;
  int b = i >> 9;
  int j = i & 511;
  const float* g = g2 + (size_t)b * (4 * H2);
  float ig = sigmoidf_(g[j]);
  float fg = sigmoidf_(g[H2 + j]);
  float gg = tanhf(g[2 * H2 + j]);
  float og = sigmoidf_(g[3 * H2 + j]);
  float cn = fg * c2[i] + ig * gg;
  c2[i] = cn;
  float h = og * tanhf(cn);
  h2_bf[i] = (bf16)h;
  size_t r = ((size_t)b * DD + t) * H2 + j;
  hs[r]    = h;
  hs_bf[r] = (bf16)h;
}

// attn2: softmax over D then weighted sum of hs -> ctx (bf16 for head GEMM)
__global__ void k_attn2(const float* __restrict__ score2, const float* __restrict__ hs,
                        bf16* __restrict__ ctx_bf)
{
  __shared__ float aw[DD];
  int b = blockIdx.x;
  if (threadIdx.x == 0) {
    const float* sc = score2 + (size_t)b * DD;
    float mx = -1e30f;
    for (int t = 0; t < DD; ++t) mx = fmaxf(mx, sc[t]);
    float sum = 0.0f;
    for (int t = 0; t < DD; ++t) { float e = __expf(sc[t] - mx); aw[t] = e; sum += e; }
    float inv = 1.0f / sum;
    for (int t = 0; t < DD; ++t) aw[t] *= inv;
  }
  __syncthreads();
  for (int j = threadIdx.x; j < H2; j += blockDim.x) {
    float acc = 0.0f;
    for (int t = 0; t < DD; ++t) acc += aw[t] * hs[((size_t)b * DD + t) * H2 + j];
    ctx_bf[(size_t)b * H2 + j] = (bf16)acc;
  }
}

// head: out[b,o] = lin4_b[o] + sum_k h3[b,k]*lin4_w[o,k]   (N=2 too small for WMMA)
__global__ void k_head(const float* __restrict__ h3, const float* __restrict__ w4,
                       const float* __restrict__ b4, float* __restrict__ out)
{
  int g = blockIdx.x * blockDim.x + threadIdx.x;
  if (g >= BN * OUTD) return;
  int b = g >> 1;
  int o = g & 1;
  float acc = b4[o];
  const float* hr = h3 + (size_t)b * FF;
  const float* wr = w4 + (size_t)o * FF;
  for (int k = 0; k < FF; ++k) acc += hr[k] * wr[k];
  out[g] = acc;
}

// ---------------------------------------------------------------------------
// Host: orchestrate the whole forward pass on `stream`.
// ---------------------------------------------------------------------------
static inline size_t align256(size_t x) { return (x + 255) & ~(size_t)255; }

extern "C" void kernel_launch(void* const* d_in, const int* in_sizes, int n_in,
                              void* d_out, int out_size, void* d_ws, size_t ws_size,
                              hipStream_t stream)
{
  (void)in_sizes; (void)n_in; (void)out_size; (void)ws_size;

  const float* sf    = (const float*)d_in[0];
  const float* tf    = (const float*)d_in[1];
  const int*   lt    = (const int*)d_in[2];
  const float* Wd_w  = (const float*)d_in[3];
  const float* Wd_b  = (const float*)d_in[4];
  const float* Wall_w= (const float*)d_in[5];
  const float* Wall_b= (const float*)d_in[6];
  const float* Uall_w= (const float*)d_in[7];
  const float* Uall_b= (const float*)d_in[8];
  const float* A1W1w = (const float*)d_in[9];
  const float* A1W1b = (const float*)d_in[10];
  const float* A1W2w = (const float*)d_in[11];
  const float* A1W2b = (const float*)d_in[12];
  const float* A1Vw  = (const float*)d_in[13];
  const float* A1Vb  = (const float*)d_in[14];
  const float* Wih   = (const float*)d_in[15];
  const float* Whh   = (const float*)d_in[16];
  const float* bih   = (const float*)d_in[17];
  const float* bhh   = (const float*)d_in[18];
  const float* A2W1w = (const float*)d_in[19];
  const float* A2W1b = (const float*)d_in[20];
  const float* A2W2w = (const float*)d_in[21];
  const float* A2W2b = (const float*)d_in[22];
  const float* A2Vw  = (const float*)d_in[23];
  const float* A2Vb  = (const float*)d_in[24];
  const float* lin3w = (const float*)d_in[25];
  const float* lin3b = (const float*)d_in[26];
  const float* lin4w = (const float*)d_in[27];
  const float* lin4b = (const float*)d_in[28];
  float* outp = (float*)d_out;

  const int NROW = DD * BN;        // 1280
  const int MROW = NROW * SS;      // 38400

  // ---- workspace carve (total ~230 MB) ----
  char* ws = (char*)d_ws;
  size_t off = 0;
  auto carve = [&](size_t bytes) -> void* {
    void* p = ws + off;
    off = align256(off + bytes);
    return p;
  };
  bf16*  xbf     = (bf16*) carve((size_t)MROW * EE * 2);
  bf16*  wcat    = (bf16*) carve((size_t)G1 * (H1 + EE) * 2);
  bf16*  wd_bf   = (bf16*) carve((size_t)H1 * H1 * 2);
  bf16*  a1w2_bf = (bf16*) carve((size_t)H1 * H1 * 2);
  bf16*  wih_bf  = (bf16*) carve((size_t)4 * H2 * H1 * 2);
  bf16*  whh_bf  = (bf16*) carve((size_t)4 * H2 * H2 * 2);
  bf16*  a2w1_bf = (bf16*) carve((size_t)H2 * H2 * 2);
  bf16*  a2w2_bf = (bf16*) carve((size_t)H2 * H2 * 2);
  bf16*  lin3_bf = (bf16*) carve((size_t)FF * H2 * 2);
  float* gbias   = (float*)carve((size_t)G1 * 4);
  float* l2bias  = (float*)carve((size_t)4 * H2 * 4);
  float* c_f32   = (float*)carve((size_t)NROW * H1 * 4);
  bf16*  c_bf    = (bf16*) carve((size_t)NROW * H1 * 2);
  bf16*  h_bf    = (bf16*) carve((size_t)NROW * H1 * 2);
  float* cs1     = (float*)carve((size_t)NROW * H1 * 4);
  float* gates   = (float*)carve((size_t)NROW * G1 * 4);
  float* outs    = (float*)carve((size_t)MROW * H1 * 4);
  bf16*  outs_bf = (bf16*) carve((size_t)MROW * H1 * 2);
  float* lastproj= (float*)carve((size_t)DD * H1 * 4);
  float* score   = (float*)carve((size_t)MROW * 4);
  float* l1      = (float*)carve((size_t)NROW * H1 * 4);
  bf16*  l1_bf   = (bf16*) carve((size_t)NROW * H1 * 2);
  float* xi      = (float*)carve((size_t)NROW * 4 * H2 * 4);
  float* c2      = (float*)carve((size_t)BN * H2 * 4);
  bf16*  h2_bf   = (bf16*) carve((size_t)BN * H2 * 2);
  float* g2      = (float*)carve((size_t)BN * 4 * H2 * 4);
  float* hs      = (float*)carve((size_t)NROW * H2 * 4);
  bf16*  hs_bf   = (bf16*) carve((size_t)NROW * H2 * 2);
  float* projhn  = (float*)carve((size_t)BN * H2 * 4);
  float* score2  = (float*)carve((size_t)NROW * 4);
  bf16*  ctx_bf  = (bf16*) carve((size_t)BN * H2 * 2);
  float* h3      = (float*)carve((size_t)BN * FF * 4);

  auto cdiv = [](int a, int b) { return (a + b - 1) / b; };
  const int TB = 256;

  // ---- precision conversion / packing ----
  { int n = MROW * EE;          k_cvt_x     <<<cdiv(n,TB),TB,0,stream>>>(xbf, sf); }
  { int n = G1 * (H1 + EE);     k_build_wcat<<<cdiv(n,TB),TB,0,stream>>>(wcat, Wall_w, Uall_w); }
  { int n = H1*H1;              k_cvt<<<cdiv(n,TB),TB,0,stream>>>(wd_bf,   Wd_w,  n); }
  { int n = H1*H1;              k_cvt<<<cdiv(n,TB),TB,0,stream>>>(a1w2_bf, A1W2w, n); }
  { int n = 4*H2*H1;            k_cvt<<<cdiv(n,TB),TB,0,stream>>>(wih_bf,  Wih,   n); }
  { int n = 4*H2*H2;            k_cvt<<<cdiv(n,TB),TB,0,stream>>>(whh_bf,  Whh,   n); }
  { int n = H2*H2;              k_cvt<<<cdiv(n,TB),TB,0,stream>>>(a2w1_bf, A2W1w, n); }
  { int n = H2*H2;              k_cvt<<<cdiv(n,TB),TB,0,stream>>>(a2w2_bf, A2W2w, n); }
  { int n = FF*H2;              k_cvt<<<cdiv(n,TB),TB,0,stream>>>(lin3_bf, lin3w, n); }
  k_add2<<<cdiv(G1,TB),TB,0,stream>>>(gbias,  Wall_b, Uall_b, G1);
  k_add2<<<cdiv(4*H2,TB),TB,0,stream>>>(l2bias, bih,  bhh,   4*H2);

  // ---- initial recurrent state = 0 ----
  k_zero_f32 <<<cdiv(NROW*H1,TB),TB,0,stream>>>(c_f32, NROW*H1);
  k_zero_bf16<<<cdiv(NROW*H1,TB),TB,0,stream>>>(c_bf,  NROW*H1);
  k_zero_bf16<<<cdiv(NROW*H1,TB),TB,0,stream>>>(h_bf,  NROW*H1);
  k_zero_f32 <<<cdiv(BN*H2,TB),TB,0,stream>>>(c2,    BN*H2);
  k_zero_bf16<<<cdiv(BN*H2,TB),TB,0,stream>>>(h2_bf, BN*H2);
  k_fill<<<cdiv(MROW,TB),TB,0,stream>>>(score,  MROW, A1Vb);
  k_fill<<<cdiv(NROW,TB),TB,0,stream>>>(score2, NROW, A2Vb);

  // ---- TimeLSTM: 30 steps, rows (d*B+b) independent ----
  for (int s = 0; s < SS; ++s) {
    // c_s1 = tanh(c @ Wd^T + Wd_b)           [1280 x 512] K=512
    k_gemm<<<dim3(H1/64, NROW/64), 128, 0, stream>>>(
        c_bf, H1, H1, nullptr, 0, 0, wd_bf, H1, Wd_b,
        nullptr, 0, cs1, H1, /*act tanh*/1, nullptr, nullptr, 1, nullptr);
    // gates_pre = [h | x_s] @ [Wall | Uall]^T + (Wall_b+Uall_b)  [1280 x 2048] K=1280
    k_gemm<<<dim3(G1/64, NROW/64), 128, 0, stream>>>(
        h_bf, H1, H1, xbf + (size_t)s * EE, SS * EE, EE, wcat, H1 + EE, gbias,
        nullptr, 0, gates, G1, 0, nullptr, nullptr, 1, nullptr);
    // elementwise TimeLSTM update; stash o-gate into outs
    k_tstep<<<cdiv(NROW*H1,TB),TB,0,stream>>>(gates, cs1, c_f32, c_bf, h_bf,
                                              outs, outs_bf, tf, s);
  }

  // ---- attn1 ----
  k_last_proj<<<cdiv(DD*H1,TB),TB,0,stream>>>(lastproj, outs, lt, A1W1w, A1W1b);
  // score += sum_n V[n]*tanh(outs@A1_W2^T + b2 + lastproj[d])   [38400 x 512]
  k_gemm<<<dim3(H1/64, MROW/64), 128, 0, stream>>>(
      outs_bf, H1, H1, nullptr, 0, 0, a1w2_bf, H1, A1W2b,
      nullptr, 0, nullptr, 0, 0, A1Vw, lastproj, BN * SS, score);
  k_attn1<<<NROW, TB, 0, stream>>>(score, outs, lt, l1, l1_bf);

  // ---- lstm2 ----
  // xi = lstm1_out @ Wih^T + (bih+bhh)      [1280 x 2048] K=512
  k_gemm<<<dim3(4*H2/64, NROW/64), 128, 0, stream>>>(
      l1_bf, H1, H1, nullptr, 0, 0, wih_bf, H1, l2bias,
      nullptr, 0, xi, 4*H2, 0, nullptr, nullptr, 1, nullptr);
  for (int t = 0; t < DD; ++t) {
    // g = h @ Whh^T + xi[:,t,:]             [256 x 2048] K=512
    k_gemm<<<dim3(4*H2/64, BN/64), 128, 0, stream>>>(
        h2_bf, H2, H2, nullptr, 0, 0, whh_bf, H2, nullptr,
        xi + (size_t)t * 4 * H2, DD * 4 * H2, g2, 4*H2, 0,
        nullptr, nullptr, 1, nullptr);
    k_l2step<<<cdiv(BN*H2,TB),TB,0,stream>>>(g2, c2, h2_bf, hs, hs_bf, t);
  }

  // ---- attn2 ----
  // proj_hn = hn2 @ A2_W1^T + b             [256 x 512]
  k_gemm<<<dim3(H2/64, BN/64), 128, 0, stream>>>(
      h2_bf, H2, H2, nullptr, 0, 0, a2w1_bf, H2, A2W1b,
      nullptr, 0, projhn, H2, 0, nullptr, nullptr, 1, nullptr);
  // score2 += sum_n V[n]*tanh(hs@A2_W2^T + b2 + proj_hn[b])  [1280 x 512]
  k_gemm<<<dim3(H2/64, NROW/64), 128, 0, stream>>>(
      hs_bf, H2, H2, nullptr, 0, 0, a2w2_bf, H2, A2W2b,
      nullptr, 0, nullptr, 0, 0, A2Vw, projhn, DD, score2);
  k_attn2<<<BN, TB, 0, stream>>>(score2, hs, ctx_bf);

  // ---- head ----
  // h3 = relu(ctx @ lin3^T + b)             [256 x 256] K=512
  k_gemm<<<dim3(FF/64, BN/64), 128, 0, stream>>>(
      ctx_bf, H2, H2, nullptr, 0, 0, lin3_bf, H2, lin3b,
      nullptr, 0, h3, FF, /*relu*/2, nullptr, nullptr, 1, nullptr);
  // out = h3 @ lin4^T + b                   [256 x 2] — tiny, plain kernel
  k_head<<<cdiv(BN*OUTD,TB),TB,0,stream>>>(h3, lin4w, lin4b, outp);
}
